// FastRCNN_90486370992182
// MI455X (gfx1250) — compile-verified, matrix operand order checked by reading, not run on hardware
//
#include <hip/hip_runtime.h>
#include <hip/hip_bf16.h>
#include <math.h>
#include <stdint.h>

typedef __attribute__((ext_vector_type(16))) _Float16 v16h;
typedef __attribute__((ext_vector_type(8)))  float    v8f;

#define NROI   128
#define RPB    64
#define C      256
#define L      49
#define NTOK   (NROI * L)        // 6272 = 392 * 16 exactly
#define HIDDEN 2048
#define CL     (C * L)           // 12544
#define NREG   84
#define NCLSO  21

// ---- WMMA fragment helpers (CDNA5 ISA 7.12.2, wave32) -----------------------
// K-offset of half-pair j (0..7) within a 16-bit A/B fragment for this lane.
__device__ __forceinline__ int frag_koff(int j, int lane) {
    return ((j & 3) << 1) + ((j >> 2) << 4) + ((lane >> 4) << 3);
}

// A matrix 16x32 f16, row-major source, rows m0..m0+15, cols k0..k0+31
__device__ __forceinline__ v16h load_a_f16(const _Float16* A, int lda, int m0, int k0, int lane) {
    v16h a;
    const _Float16* base = A + (size_t)(m0 + (lane & 15)) * lda + k0;
#pragma unroll
    for (int j = 0; j < 8; ++j) {
        int ko = frag_koff(j, lane);
        a[2*j]   = base[ko];
        a[2*j+1] = base[ko + 1];
    }
    return a;
}

// A fragment with row bound (rows >= rmax read as 0)
__device__ __forceinline__ v16h load_a_f16_bnd(const _Float16* A, int lda, int m0, int k0,
                                               int lane, int rmax) {
    v16h a;
    int row = m0 + (lane & 15);
    bool ok = row < rmax;
    const _Float16* base = A + (size_t)(ok ? row : 0) * lda + k0;
#pragma unroll
    for (int j = 0; j < 8; ++j) {
        int ko = frag_koff(j, lane);
        a[2*j]   = ok ? base[ko]     : (_Float16)0.0f;
        a[2*j+1] = ok ? base[ko + 1] : (_Float16)0.0f;
    }
    return a;
}

// B matrix 32x16 from f32 row-major [K x N], converted to f16 in-register
// (weights stay f32 in HBM -> no extra memory traffic).
__device__ __forceinline__ v16h load_b_f32(const float* B, int ldb, int k0, int n0, int lane) {
    v16h b;
    int col = n0 + (lane & 15);
#pragma unroll
    for (int j = 0; j < 8; ++j) {
        int ko = frag_koff(j, lane);
        b[2*j]   = (_Float16)B[(size_t)(k0 + ko)     * ldb + col];
        b[2*j+1] = (_Float16)B[(size_t)(k0 + ko + 1) * ldb + col];
    }
    return b;
}

// B fragment from an f32 [32 x 16] tile staged in LDS
__device__ __forceinline__ v16h load_b_lds_f32(const float* Bt, int lane) {
    v16h b;
    int col = lane & 15;
#pragma unroll
    for (int j = 0; j < 8; ++j) {
        int ko = frag_koff(j, lane);
        b[2*j]   = (_Float16)Bt[ko * 16 + col];
        b[2*j+1] = (_Float16)Bt[(ko + 1) * 16 + col];
    }
    return b;
}

// B matrix 32x16 from f16 row-major [K x N] with K-row bound (rows >= kmax -> 0)
__device__ __forceinline__ v16h load_b_f16_bnd(const _Float16* B, int ldb, int k0, int n0,
                                               int lane, int kmax) {
    v16h b;
    int col = n0 + (lane & 15);
#pragma unroll
    for (int j = 0; j < 8; ++j) {
        int ko = frag_koff(j, lane);
        int r0 = k0 + ko, r1 = r0 + 1;
        b[2*j]   = (r0 < kmax) ? B[(size_t)r0 * ldb + col] : (_Float16)0.0f;
        b[2*j+1] = (r1 < kmax) ? B[(size_t)r1 * ldb + col] : (_Float16)0.0f;
    }
    return b;
}

// B = M^T where M is f16 row-major [N x K] (for Q @ K^T). Col bound on N.
__device__ __forceinline__ v16h load_bt_f16_bnd(const _Float16* M, int ldm, int k0, int n0,
                                                int lane, int nmax) {
    v16h b;
    int col = n0 + (lane & 15);
    bool ok = col < nmax;
    const _Float16* base = M + (size_t)(ok ? col : 0) * ldm + k0;
#pragma unroll
    for (int j = 0; j < 8; ++j) {
        int ko = frag_koff(j, lane);
        b[2*j]   = ok ? base[ko]     : (_Float16)0.0f;
        b[2*j+1] = ok ? base[ko + 1] : (_Float16)0.0f;
    }
    return b;
}

__device__ __forceinline__ v8f wmma_f16(v16h a, v16h b, v8f c) {
    return __builtin_amdgcn_wmma_f32_16x16x32_f16(false, a, false, b, (short)0, c, false, false);
}

// C/D 16x16 f32 layout: VGPR i -> row m0 + i + 8*(lane>>4), col n0 + (lane&15)
__device__ __forceinline__ int crow(int i, int lane) { return i + ((lane >> 4) << 3); }

// ---- Kernel 1: ROI level-select + expand + max-pool -------------------------
__global__ void pool_kernel(const float* __restrict__ rois,
                            const float* __restrict__ fm0, const float* __restrict__ fm1,
                            const float* __restrict__ fm2, const float* __restrict__ fm3,
                            float* __restrict__ pooled,      // [NROI, L, C]
                            _Float16* __restrict__ tokf16) { // [NTOK, C]
    int n = blockIdx.x;      // ROI
    int c = threadIdx.x;     // channel
    int b = n / RPB;
    const float* r = rois + (size_t)n * 4;
    float x1f = r[0], y1f = r[1], x2f = r[2], y2f = r[3];
    float size = sqrtf((x2f - x1f) * (y2f - y1f));
    int lvl = (int)(3.0f + logf(size / 224.0f));   // trunc toward zero, matches ref
    lvl = min(max(lvl, 0), 3);
    const float strides[4] = {4.0f, 8.0f, 16.0f, 32.0f};
    const int   Hs[4]      = {160, 80, 40, 20};
    const float* fms[4]    = {fm0, fm1, fm2, fm3};
    float s = strides[lvl];
    int Hl = Hs[lvl];
    int x1 = (int)(x1f / s), y1 = (int)(y1f / s);
    int x2 = (int)(x2f / s), y2 = (int)(y2f / s);
#pragma unroll
    for (int it = 0; it < 7; ++it) {
        if (y2 - y1 + 1 < 7) { y1 = max(y1 - 1, 0); y2 = min(y2 + 1, Hl - 1); }
    }
#pragma unroll
    for (int it = 0; it < 7; ++it) {
        if (x2 - x1 + 1 < 7) { x1 = max(x1 - 1, 0); x2 = min(x2 + 1, Hl - 1); }
    }
    int rh = y2 - y1 + 1, rw = x2 - x1 + 1;
    int kh = (rh + 6) / 7, sh = rh / 7;
    int kw = (rw + 6) / 7, sw = rw / 7;
    const float* f = fms[lvl] + ((size_t)b * C + c) * (size_t)Hl * Hl;
    for (int ph = 0; ph < 7; ++ph) {
        for (int pw = 0; pw < 7; ++pw) {
            float m = -3.402823466e38f;
            for (int k = 0; k < kh; ++k) {
                int row = min(y1 + ph * sh + k, Hl - 1);
                const float* fr = f + (size_t)row * Hl;
                for (int kk = 0; kk < kw; ++kk) {
                    int col = min(x1 + pw * sw + kk, Hl - 1);
                    m = fmaxf(m, fr[col]);
                }
            }
            int l = ph * 7 + pw;
            size_t o = ((size_t)n * L + l) * C + c;
            pooled[o] = m;
            tokf16[o] = (_Float16)m;
        }
    }
}

// ---- Kernel 2: generic tile GEMM (A f16 [M,Kd], W f32 [Kd,N]) -> f16 out ----
__global__ void gemm_tile(const _Float16* __restrict__ A, const float* __restrict__ W,
                          const float* __restrict__ bias, _Float16* __restrict__ Out,
                          int Kd, int Nd) {
    int m0 = blockIdx.x * 16, n0 = blockIdx.y * 16;
    int lane = threadIdx.x;
    v8f acc = {};
    for (int k0 = 0; k0 < Kd; k0 += 32) {
        if (k0 + 32 < Kd)   // prefetch next weight stripe (global_prefetch_b8)
            __builtin_prefetch(&W[(size_t)(k0 + 32 + (lane & 31)) * Nd + n0], 0, 1);
        v16h a = load_a_f16(A, Kd, m0, k0, lane);
        v16h b = load_b_f32(W, Nd, k0, n0, lane);
        acc = wmma_f16(a, b, acc);
    }
    int col = n0 + (lane & 15);
    float bv = bias[col];
#pragma unroll
    for (int i = 0; i < 8; ++i)
        Out[(size_t)(m0 + crow(i, lane)) * Nd + col] = (_Float16)(acc[i] + bv);
}

// ---- Kernel 3: per-ROI attention (scores + softmax + P@V), WMMA + LDS -------
__global__ void attn_kernel(const _Float16* __restrict__ Q, const _Float16* __restrict__ Km,
                            const _Float16* __restrict__ V, _Float16* __restrict__ CTX) {
    __shared__ float    S[64][64];
    __shared__ _Float16 P[64][64];
    int n = blockIdx.x;
    int tid = threadIdx.x, wave = tid >> 5, lane = tid & 31;
    const _Float16* q = Q  + (size_t)n * L * C;
    const _Float16* k = Km + (size_t)n * L * C;
    const _Float16* v = V  + (size_t)n * L * C;

    // S = q @ k^T : 4x4 tiles of 16x16, K=256
    for (int t = wave; t < 16; t += 8) {
        int mt = t >> 2, nt = t & 3;
        v8f acc = {};
        for (int k0 = 0; k0 < C; k0 += 32) {
            v16h af = load_a_f16_bnd(q, C, mt * 16, k0, lane, L);
            v16h bf = load_bt_f16_bnd(k, C, k0, nt * 16, lane, L);
            acc = wmma_f16(af, bf, acc);
        }
#pragma unroll
        for (int i = 0; i < 8; ++i)
            S[mt * 16 + crow(i, lane)][nt * 16 + (lane & 15)] = acc[i];
    }
    __syncthreads();

    // softmax over 49 valid columns, scale = 1/16 (sqrt(256))
    if (tid < 64) {
        const float inv_scale = 1.0f / 16.0f;
        float mx = -3.402823466e38f;
        for (int j = 0; j < L; ++j) mx = fmaxf(mx, S[tid][j] * inv_scale);
        float sum = 0.0f;
        for (int j = 0; j < L; ++j) {
            float e = __expf(S[tid][j] * inv_scale - mx);
            sum += e;
            S[tid][j] = e;
        }
        float inv = 1.0f / sum;
        for (int j = 0; j < 64; ++j)
            P[tid][j] = (j < L) ? (_Float16)(S[tid][j] * inv) : (_Float16)0.0f;
    }
    __syncthreads();

    // ctx = P @ v : 4x16 tiles, K=64
    for (int t = wave; t < 64; t += 8) {
        int mt = t >> 4, nt = t & 15;
        v8f acc = {};
        for (int k0 = 0; k0 < 64; k0 += 32) {
            v16h af = load_a_f16(&P[0][0], 64, mt * 16, k0, lane);
            v16h bf = load_b_f16_bnd(v, C, k0, nt * 16, lane, L);
            acc = wmma_f16(af, bf, acc);
        }
#pragma unroll
        for (int i = 0; i < 8; ++i) {
            int row = mt * 16 + crow(i, lane);
            if (row < L)
                CTX[((size_t)n * L + row) * C + nt * 16 + (lane & 15)] = (_Float16)acc[i];
        }
    }
}

// ---- Kernel 4: out-projection + bias + residual, scatter to [N, C*L] f16 ----
__global__ void outproj_kernel(const _Float16* __restrict__ CTX, const float* __restrict__ wo,
                               const float* __restrict__ bo, const float* __restrict__ pooled,
                               _Float16* __restrict__ Xh) {
    int m0 = blockIdx.x * 16, n0 = blockIdx.y * 16;
    int lane = threadIdx.x;
    v8f acc = {};
    for (int k0 = 0; k0 < C; k0 += 32) {
        v16h a = load_a_f16(CTX, C, m0, k0, lane);
        v16h b = load_b_f32(wo, C, k0, n0, lane);
        acc = wmma_f16(a, b, acc);
    }
    int col = n0 + (lane & 15);
    float bv = bo[col];
#pragma unroll
    for (int i = 0; i < 8; ++i) {
        int g = m0 + crow(i, lane);          // global token index
        int n = g / L, l = g % L;
        float val = acc[i] + bv + pooled[(size_t)g * C + col];
        Xh[(size_t)n * CL + col * L + l] = (_Float16)val;
    }
}

// ---- Kernel 5: FC layer, M=128 (8 waves = 8 m-tiles), grid = n-tiles --------
// B tile [32 x 16] f32 is staged global->LDS with CDNA5 async-to-LDS copies
// (ASYNCcnt), double-buffered so the next tile streams while WMMA consumes
// the current one. Each block reads its 16-column weight stripe exactly once.
__global__ void fc_kernel(const _Float16* __restrict__ A, const float* __restrict__ W,
                          const float* __restrict__ bias, _Float16* __restrict__ Out,
                          int Kd, int Nd) {
    __shared__ float Bt[2][32 * 16];
    int nt = blockIdx.x;
    int tid = threadIdx.x;
    int wave = tid >> 5, lane = tid & 31;
    int m0 = wave * 16, n0 = nt * 16;

    // staging coords: 512 floats per tile, 256 threads x one b64 async copy
    int r0 = tid >> 3;             // 0..31 (k row within tile)
    int cp = (tid & 7) * 2;        // 0,2,..,14 (column pair)
    unsigned lds0 = (unsigned)(uintptr_t)&Bt[0][r0 * 16 + cp];
    unsigned lds1 = (unsigned)(uintptr_t)&Bt[1][r0 * 16 + cp];

    // issue async copy of tile at k-offset k0 into buffer buf
    #define FC_ISSUE(k0_, ldsoff_)                                              \
        do {                                                                    \
            unsigned goff = (unsigned)((((k0_) + r0) * Nd + n0 + cp) * 4);      \
            asm volatile("global_load_async_to_lds_b64 %0, %1, %2 offset:0"     \
                         :: "v"(ldsoff_), "v"(goff), "s"(W) : "memory");        \
        } while (0)

    FC_ISSUE(0, lds0);
    v8f acc = {};
    for (int k0 = 0; k0 < Kd; k0 += 32) {
        int cur = (k0 >> 5) & 1;
        if (k0 + 32 < Kd) {
            FC_ISSUE(k0 + 32, cur ? lds0 : lds1);
            asm volatile("s_wait_asynccnt 0x1" ::: "memory");  // cur tile landed
        } else {
            asm volatile("s_wait_asynccnt 0x0" ::: "memory");  // drain
        }
        __syncthreads();                      // all waves' stages visible
        v16h a = load_a_f16(A, Kd, m0, k0, lane);
        v16h b = load_b_lds_f32(Bt[cur], lane);
        acc = wmma_f16(a, b, acc);
        __syncthreads();                      // reads done before buf reuse
    }
    #undef FC_ISSUE

    int col = n0 + (lane & 15);
    float bv = bias[col];
#pragma unroll
    for (int i = 0; i < 8; ++i) {
        float vv = fmaxf(acc[i] + bv, 0.0f);   // ReLU
        Out[(size_t)(m0 + crow(i, lane)) * Nd + col] = (_Float16)vv;
    }
}

// ---- Kernel 6: bbox-reg head (scalar dot, 27 MFLOP total) -------------------
__global__ void reg_head(const _Float16* __restrict__ H2, const float* __restrict__ wreg,
                         const float* __restrict__ breg, float* __restrict__ out) {
    int idx = blockIdx.x * blockDim.x + threadIdx.x;
    if (idx >= NROI * NREG) return;
    int n = idx / NREG, j = idx % NREG;
    const _Float16* h = H2 + (size_t)n * HIDDEN;
    float s = breg[j];
    for (int kk = 0; kk < HIDDEN; ++kk) s += (float)h[kk] * wreg[(size_t)kk * NREG + j];
    out[idx] = s;
}

// ---- Kernel 7: class head + softmax ----------------------------------------
__global__ void cls_head(const _Float16* __restrict__ H2, const float* __restrict__ wcls,
                         const float* __restrict__ bcls, float* __restrict__ out) {
    __shared__ float logit[NCLSO];
    int n = blockIdx.x, t = threadIdx.x;
    if (t < NCLSO) {
        const _Float16* h = H2 + (size_t)n * HIDDEN;
        float s = bcls[t];
        for (int kk = 0; kk < HIDDEN; ++kk) s += (float)h[kk] * wcls[(size_t)kk * NCLSO + t];
        logit[t] = s;
    }
    __syncthreads();
    if (t < NCLSO) {
        float mx = -3.402823466e38f;
        for (int j = 0; j < NCLSO; ++j) mx = fmaxf(mx, logit[j]);
        float sum = 0.0f;
        for (int j = 0; j < NCLSO; ++j) sum += __expf(logit[j] - mx);
        out[NROI * NREG + n * NCLSO + t] = __expf(logit[t] - mx) / sum;
    }
}

// ---- launch -----------------------------------------------------------------
extern "C" void kernel_launch(void* const* d_in, const int* in_sizes, int n_in,
                              void* d_out, int out_size, void* d_ws, size_t ws_size,
                              hipStream_t stream) {
    const float* rois = (const float*)d_in[0];
    const float* fm0  = (const float*)d_in[1];
    const float* fm1  = (const float*)d_in[2];
    const float* fm2  = (const float*)d_in[3];
    const float* fm3  = (const float*)d_in[4];
    const float* wq = (const float*)d_in[5];  const float* bq = (const float*)d_in[6];
    const float* wk = (const float*)d_in[7];  const float* bk = (const float*)d_in[8];
    const float* wv = (const float*)d_in[9];  const float* bv = (const float*)d_in[10];
    const float* wo = (const float*)d_in[11]; const float* bo = (const float*)d_in[12];
    const float* w1 = (const float*)d_in[13]; const float* b1 = (const float*)d_in[14];
    const float* w2 = (const float*)d_in[15]; const float* b2 = (const float*)d_in[16];
    const float* wreg = (const float*)d_in[17]; const float* breg = (const float*)d_in[18];
    const float* wcls = (const float*)d_in[19]; const float* bcls = (const float*)d_in[20];
    float* out = (float*)d_out;

    // workspace carve-up (~27 MB)
    char* p = (char*)d_ws;
    float*    pooled = (float*)p;       p += (size_t)NTOK * C * sizeof(float);
    _Float16* tok    = (_Float16*)p;    p += (size_t)NTOK * C * sizeof(_Float16);
    _Float16* Qb     = (_Float16*)p;    p += (size_t)NTOK * C * sizeof(_Float16);
    _Float16* Kb     = (_Float16*)p;    p += (size_t)NTOK * C * sizeof(_Float16);
    _Float16* Vb     = (_Float16*)p;    p += (size_t)NTOK * C * sizeof(_Float16);
    _Float16* CTX    = (_Float16*)p;    p += (size_t)NTOK * C * sizeof(_Float16);
    _Float16* Xh     = (_Float16*)p;    p += (size_t)NROI * CL * sizeof(_Float16);
    _Float16* H1     = (_Float16*)p;    p += (size_t)NROI * HIDDEN * sizeof(_Float16);
    _Float16* H2     = (_Float16*)p;    p += (size_t)NROI * HIDDEN * sizeof(_Float16);

    pool_kernel<<<NROI, C, 0, stream>>>(rois, fm0, fm1, fm2, fm3, pooled, tok);

    dim3 gq(NTOK / 16, C / 16);   // 392 x 16
    gemm_tile<<<gq, 32, 0, stream>>>(tok, wq, bq, Qb, C, C);
    gemm_tile<<<gq, 32, 0, stream>>>(tok, wk, bk, Kb, C, C);
    gemm_tile<<<gq, 32, 0, stream>>>(tok, wv, bv, Vb, C, C);

    attn_kernel<<<NROI, 256, 0, stream>>>(Qb, Kb, Vb, CTX);

    outproj_kernel<<<gq, 32, 0, stream>>>(CTX, wo, bo, pooled, Xh);

    fc_kernel<<<HIDDEN / 16, 256, 0, stream>>>(Xh, w1, b1, H1, CL, HIDDEN);
    fc_kernel<<<HIDDEN / 16, 256, 0, stream>>>(H1, w2, b2, H2, HIDDEN, HIDDEN);

    reg_head<<<(NROI * NREG + 255) / 256, 256, 0, stream>>>(H2, wreg, breg, out);
    cls_head<<<NROI, 32, 0, stream>>>(H2, wcls, bcls, out);
}
